// RPN_23356032156251
// MI455X (gfx1250) — matrix operand order repaired and checked by reading, not compiled, not run on hardware
//
#include <hip/hip_runtime.h>
#include <hip/hip_bf16.h>

#define BATCH 8
#define CIN 512
#define FH 50
#define FW 76
#define PH 52                  // padded height (halo 1)
#define PW 78                  // padded width  (halo 1)
#define HWPIX (FH*FW)          // 3800
#define MROWS (BATCH*HWPIX)    // 30400
#define NPIX (HWPIX*9)         // 34200
#define PRE_NMS 6000
#define POST_NMS 300
#define NWORDS 188             // ceil(6000/32)
#define SORTN 65536
#define NMS_THRESH 0.7f

typedef __attribute__((ext_vector_type(16))) _Float16 v16h;
typedef __attribute__((ext_vector_type(8)))  _Float16 v8h;
typedef __attribute__((ext_vector_type(8)))  float    v8f;

union AFrag { v16h v; v8h h[2]; };

static __device__ inline v8f wmma_f16(v16h a, v16h b, v8f c) {
  // D = A(16x32 f16) * B(32x16 f16) + C(16x16 f32)
  return __builtin_amdgcn_wmma_f32_16x16x32_f16(
      /*neg_a=*/false, a, /*neg_b=*/false, b,
      /*c_mod=*/(short)0, c, /*reuse_a=*/false, /*reuse_b=*/false);
}

// ---------------------------------------------------------------------------
// 1) NCHW f32 -> zero-halo-padded NHWC f16 ([B][PH][PW][C])
// ---------------------------------------------------------------------------
__global__ void k_x_to_nhwc(const float* __restrict__ src, _Float16* __restrict__ dst) {
  int t = blockIdx.x * blockDim.x + threadIdx.x;
  if (t >= BATCH * PH * PW * CIN) return;
  int c = t % CIN;
  int rest = t / CIN;
  int xp = rest % PW; rest /= PW;
  int yp = rest % PH;
  int b  = rest / PH;
  float v = 0.0f;
  if (yp >= 1 && yp <= FH && xp >= 1 && xp <= FW)
    v = src[(((size_t)b * CIN + c) * FH + (yp - 1)) * FW + (xp - 1)];
  dst[t] = (_Float16)v;
}

// ---------------------------------------------------------------------------
// 2) Pack conv1 weights (OIHW f32) into per-tap WMMA B-fragment order (f16).
//    Fragment f = (kykx*16 + kt)*32 + ntile; per lane 16 contiguous f16.
//    B layout: lane l holds col n = l%16; element i -> k = (l>=16?8:0) + (i<8 ? i : 8+i)
// ---------------------------------------------------------------------------
__global__ void k_pack_w1(const float* __restrict__ w, _Float16* __restrict__ wpack) {
  int t = blockIdx.x * blockDim.x + threadIdx.x;
  if (t >= 9 * 16 * 32 * 32) return;
  int l = t & 31;
  int f = t >> 5;
  int nt = f & 31;
  int kt = (f >> 5) & 15;
  int kykx = f >> 9;
  int ky = kykx / 3, kx = kykx % 3;
  int oc = nt * 16 + (l & 15);
  int kb = (l >= 16) ? 16 : 0;
  _Float16* dst = wpack + (size_t)t * 16;
#pragma unroll
  for (int i = 0; i < 16; ++i) {
    int ic = kt * 32 + (kb >> 1) + (i < 8 ? i : 8 + i);
    dst[i] = (_Float16)w[(((size_t)oc * CIN + ic) * 3 + ky) * 3 + kx];
  }
}

// ---------------------------------------------------------------------------
// 3) Pack loc(36)+score(18) 1x1 weights into one [K=512 x N=64] B matrix.
// ---------------------------------------------------------------------------
__global__ void k_pack_whead(const float* __restrict__ loc_w,
                             const float* __restrict__ score_w,
                             _Float16* __restrict__ hpack) {
  int t = blockIdx.x * blockDim.x + threadIdx.x;
  if (t >= 16 * 4 * 32) return;
  int l = t & 31;
  int f = t >> 5;
  int nt = f & 3;
  int kt = f >> 2;
  int n = nt * 16 + (l & 15);
  int kb = (l >= 16) ? 16 : 0;
  _Float16* dst = hpack + (size_t)t * 16;
#pragma unroll
  for (int i = 0; i < 16; ++i) {
    int ic = kt * 32 + (kb >> 1) + (i < 8 ? i : 8 + i);
    float v = 0.0f;
    if (n < 36)      v = loc_w[(size_t)n * CIN + ic];
    else if (n < 54) v = score_w[(size_t)(n - 36) * CIN + ic];
    dst[i] = (_Float16)v;
  }
}

// ---------------------------------------------------------------------------
// 4) conv1 3x3 as implicit GEMM over the halo-padded input. Block: 256 thr =
//    8 waves (2 M-waves x 4 N-waves); block tile 64x256; wave tile 32x64
//    (2x4 WMMA tiles). All 9 taps are constant offsets off one base pointer.
//    Fused bias + ReLU, feat stored f16 NHWC (unpadded).
// ---------------------------------------------------------------------------
__global__ __launch_bounds__(256) void k_conv1(const _Float16* __restrict__ xb,
                                               const _Float16* __restrict__ wpack,
                                               const float* __restrict__ bias,
                                               _Float16* __restrict__ feat) {
  const int tid = threadIdx.x;
  const int l   = tid & 31;
  const int wv  = tid >> 5;
  const int mw  = wv >> 2;          // 0..1
  const int nw  = wv & 3;           // 0..3
  const int lr  = l & 15;
  const int hi  = l >> 4;
  const int abase = hi * 8;         // A fragment per-lane K offset

  const int Mstart = blockIdx.x * 64 + mw * 32;
  const int nt0    = blockIdx.y * 16 + nw * 4;  // first of 4 n-tiles

  // Per-lane base pointer into padded input at tap (ky=0,kx=0).
  const _Float16* ap[2];
#pragma unroll
  for (int mf = 0; mf < 2; ++mf) {
    int r = Mstart + mf * 16 + lr;
    int bb = r / HWPIX;
    int p  = r % HWPIX;
    int py = p / FW;
    int px = p % FW;
    ap[mf] = xb + (((size_t)bb * PH + py) * PW + px) * CIN + abase;
  }

  v8f zz = {};
  v8f acc[2][4] = {{zz, zz, zz, zz}, {zz, zz, zz, zz}};

  for (int kykx = 0; kykx < 9; ++kykx) {
    const int tap = ((kykx / 3) * PW + (kykx % 3)) * CIN;  // constant elem offset
    for (int kt = 0; kt < 16; ++kt) {
      v16h a[2];
#pragma unroll
      for (int mf = 0; mf < 2; ++mf) {
        AFrag f;
        f.h[0] = *(const v8h*)(ap[mf] + tap + kt * 32);
        f.h[1] = *(const v8h*)(ap[mf] + tap + kt * 32 + 16);
        a[mf] = f.v;
      }
      v16h bfr[4];
#pragma unroll
      for (int nf = 0; nf < 4; ++nf) {
        size_t fidx = (size_t)(kykx * 16 + kt) * 32 + (nt0 + nf);
        bfr[nf] = *(const v16h*)(wpack + (fidx * 32 + l) * 16);
      }
#pragma unroll
      for (int mf = 0; mf < 2; ++mf)
#pragma unroll
        for (int nf = 0; nf < 4; ++nf)
          acc[mf][nf] = wmma_f16(a[mf], bfr[nf], acc[mf][nf]);
    }
  }

#pragma unroll
  for (int mf = 0; mf < 2; ++mf)
#pragma unroll
    for (int nf = 0; nf < 4; ++nf)
#pragma unroll
      for (int v = 0; v < 8; ++v) {
        int row = Mstart + mf * 16 + v + hi * 8;
        int col = (nt0 + nf) * 16 + lr;
        float val = acc[mf][nf][v] + bias[col];
        val = val > 0.0f ? val : 0.0f;
        feat[(size_t)row * CIN + col] = (_Float16)val;
      }
}

// ---------------------------------------------------------------------------
// 5) Head GEMM: [M x 512] x [512 x 64] via WMMA; scatter locs/scores to d_out.
//    Block: 128 thr = 4 waves; wave tile 16x64.
// ---------------------------------------------------------------------------
__global__ __launch_bounds__(128) void k_head(const _Float16* __restrict__ feat,
                                              const _Float16* __restrict__ hpack,
                                              const float* __restrict__ loc_b,
                                              const float* __restrict__ score_b,
                                              float* __restrict__ out_locs,
                                              float* __restrict__ out_scores) {
  const int tid = threadIdx.x;
  const int l   = tid & 31;
  const int wv  = tid >> 5;
  const int lr  = l & 15;
  const int hi  = l >> 4;
  const int abase = hi * 8;
  const int Mstart = blockIdx.x * 64 + wv * 16;

  v8f zz = {};
  v8f acc[4] = {zz, zz, zz, zz};

  const _Float16* ap = feat + (size_t)(Mstart + lr) * CIN + abase;
  for (int kt = 0; kt < 16; ++kt) {
    AFrag f;
    f.h[0] = *(const v8h*)(ap + kt * 32);
    f.h[1] = *(const v8h*)(ap + kt * 32 + 16);
#pragma unroll
    for (int nf = 0; nf < 4; ++nf) {
      v16h b = *(const v16h*)(hpack + (((size_t)kt * 4 + nf) * 32 + l) * 16);
      acc[nf] = wmma_f16(f.v, b, acc[nf]);
    }
  }

#pragma unroll
  for (int nf = 0; nf < 4; ++nf)
#pragma unroll
    for (int v = 0; v < 8; ++v) {
      int n = nf * 16 + lr;
      int m = Mstart + v + hi * 8;
      int b = m / HWPIX, pix = m % HWPIX;
      float val = acc[nf][v];
      if (n < 36) {
        int a = n >> 2, d = n & 3;
        out_locs[(((size_t)b * NPIX) + (size_t)pix * 9 + a) * 4 + d] = val + loc_b[n];
      } else if (n < 54) {
        int c = n - 36;
        int a = c >> 1, j = c & 1;
        out_scores[(((size_t)b * NPIX) + (size_t)pix * 9 + a) * 2 + j] = val + score_b[c];
      }
    }
}

// ---------------------------------------------------------------------------
// 6) Anchors + softmax-fg + loc2bbox + clip + min-size; build sort keys.
//    key = (~orderable(score)) << 32 | index  -> ascending sort == top_k order
// ---------------------------------------------------------------------------
__global__ void k_boxes(const float* __restrict__ locs, const float* __restrict__ scores,
                        const int* __restrict__ p_imgh, const int* __restrict__ p_imgw,
                        float* __restrict__ boxes, float* __restrict__ score_ws,
                        unsigned long long* __restrict__ keys,
                        float* __restrict__ out_anchor) {
  int t = blockIdx.x * blockDim.x + threadIdx.x;
  if (t >= BATCH * SORTN) return;
  int b = t >> 16;
  int i = t & (SORTN - 1);
  if (i >= NPIX) { keys[t] = ~0ull; return; }

  const float ratios[3] = {0.5f, 1.0f, 2.0f};
  const float scales[3] = {8.0f, 16.0f, 32.0f};
  int a = i % 9, pix = i / 9;
  float r = ratios[a / 3], s = scales[a % 3];
  float ha = 16.0f * s * sqrtf(r);
  float wa = 16.0f * s * sqrtf(1.0f / r);
  float cy0 = (float)(pix / FW) * 16.0f + 8.0f;
  float cx0 = (float)(pix % FW) * 16.0f + 8.0f;
  float ax1 = cx0 - 0.5f * wa, ay1 = cy0 - 0.5f * ha;
  float ax2 = cx0 + 0.5f * wa, ay2 = cy0 + 0.5f * ha;
  if (b == 0) {
    out_anchor[(size_t)i * 4 + 0] = ax1;
    out_anchor[(size_t)i * 4 + 1] = ay1;
    out_anchor[(size_t)i * 4 + 2] = ax2;
    out_anchor[(size_t)i * 4 + 3] = ay2;
  }

  const float* lp = locs + ((size_t)b * NPIX + i) * 4;
  float cx = lp[0] * wa + cx0;
  float cy = lp[1] * ha + cy0;
  float w  = __expf(lp[2]) * wa;
  float h  = __expf(lp[3]) * ha;
  float img_w = (float)p_imgw[0], img_h = (float)p_imgh[0];
  float x1 = fminf(fmaxf(cx - 0.5f * w, 0.0f), img_w);
  float y1 = fminf(fmaxf(cy - 0.5f * h, 0.0f), img_h);
  float x2 = fminf(fmaxf(cx + 0.5f * w, 0.0f), img_w);
  float y2 = fminf(fmaxf(cy + 0.5f * h, 0.0f), img_h);

  float* bp = boxes + ((size_t)b * NPIX + i) * 4;
  bp[0] = x1; bp[1] = y1; bp[2] = x2; bp[3] = y2;

  float s0 = scores[((size_t)b * NPIX + i) * 2 + 0];
  float s1 = scores[((size_t)b * NPIX + i) * 2 + 1];
  float mx = fmaxf(s0, s1);
  float e0 = __expf(s0 - mx), e1 = __expf(s1 - mx);
  float fg = e1 / (e0 + e1);

  bool valid = ((x2 - x1) >= 16.0f) && ((y2 - y1) >= 16.0f);
  float sc = valid ? fg : -__builtin_inff();
  score_ws[(size_t)b * NPIX + i] = sc;

  unsigned u = __float_as_uint(sc);
  u = (u & 0x80000000u) ? ~u : (u | 0x80000000u);   // orderable: ascending == ascending float
  keys[t] = ((unsigned long long)(~u) << 32) | (unsigned)i;
}

// ---------------------------------------------------------------------------
// 7) Bitonic sort step (ascending) on [BATCH][65536] 64-bit keys.
// ---------------------------------------------------------------------------
__global__ void k_bitonic(unsigned long long* __restrict__ keys, int k, int j) {
  int t = blockIdx.x * blockDim.x + threadIdx.x;
  if (t >= BATCH * SORTN) return;
  int b  = t >> 16;
  int id = t & (SORTN - 1);
  int ixj = id ^ j;
  if (ixj <= id) return;
  unsigned long long* base = keys + ((size_t)b << 16);
  unsigned long long a0 = base[id], a1 = base[ixj];
  bool up = ((id & k) == 0);
  if (up ? (a0 > a1) : (a0 < a1)) { base[id] = a1; base[ixj] = a0; }
}

// ---------------------------------------------------------------------------
// 8) Gather top-6000 boxes/areas/scores in sorted order.
// ---------------------------------------------------------------------------
__global__ void k_gather(const unsigned long long* __restrict__ keys,
                         const float* __restrict__ boxes, const float* __restrict__ score_ws,
                         float* __restrict__ sboxes, float* __restrict__ sarea,
                         float* __restrict__ sscore) {
  int t = blockIdx.x * blockDim.x + threadIdx.x;
  if (t >= BATCH * PRE_NMS) return;
  int b = t / PRE_NMS;
  unsigned long long key = keys[((size_t)b << 16) + (t % PRE_NMS)];
  int idx = (int)(unsigned)(key & 0xffffffffu);
  const float* bp = boxes + ((size_t)b * NPIX + idx) * 4;
  float x1 = bp[0], y1 = bp[1], x2 = bp[2], y2 = bp[3];
  sboxes[(size_t)t * 4 + 0] = x1;
  sboxes[(size_t)t * 4 + 1] = y1;
  sboxes[(size_t)t * 4 + 2] = x2;
  sboxes[(size_t)t * 4 + 3] = y2;
  sarea[t]  = (x2 - x1) * (y2 - y1);
  sscore[t] = score_ws[(size_t)b * NPIX + idx];
}

// ---------------------------------------------------------------------------
// 9) IoU suppression bitmask: mask[b][i][cb] bit jj set iff IoU(i, cb*32+jj)>T
//    and (cb*32+jj) > i.
// ---------------------------------------------------------------------------
__global__ __launch_bounds__(128) void k_nms_mask(const float* __restrict__ sboxes,
                                                  const float* __restrict__ sarea,
                                                  unsigned* __restrict__ mask) {
  int b  = blockIdx.z;
  int i  = blockIdx.x * 128 + threadIdx.x;
  int cb = blockIdx.y;
  __shared__ float sx1[32], sy1[32], sx2[32], sy2[32], sa[32];
  int tx = threadIdx.x;
  if (tx < 32) {
    int j = cb * 32 + tx;
    if (j < PRE_NMS) {
      const float* p = sboxes + ((size_t)b * PRE_NMS + j) * 4;
      sx1[tx] = p[0]; sy1[tx] = p[1]; sx2[tx] = p[2]; sy2[tx] = p[3];
      sa[tx]  = sarea[(size_t)b * PRE_NMS + j];
    } else {
      sx1[tx] = 0; sy1[tx] = 0; sx2[tx] = 0; sy2[tx] = 0; sa[tx] = 0;
    }
  }
  __syncthreads();
  if (i >= PRE_NMS) return;
  const float* p = sboxes + ((size_t)b * PRE_NMS + i) * 4;
  float x1 = p[0], y1 = p[1], x2 = p[2], y2 = p[3];
  float ai = sarea[(size_t)b * PRE_NMS + i];
  unsigned word = 0;
  for (int jj = 0; jj < 32; ++jj) {
    int j = cb * 32 + jj;
    if (j < PRE_NMS && j > i) {
      float xx1 = fmaxf(x1, sx1[jj]);
      float yy1 = fmaxf(y1, sy1[jj]);
      float xx2 = fminf(x2, sx2[jj]);
      float yy2 = fminf(y2, sy2[jj]);
      float inter = fmaxf(xx2 - xx1, 0.0f) * fmaxf(yy2 - yy1, 0.0f);
      float iou = inter / (ai + sa[jj] - inter + 1e-9f);
      if (iou > NMS_THRESH) word |= (1u << jj);
    }
  }
  mask[((size_t)b * PRE_NMS + i) * NWORDS + cb] = word;
}

// ---------------------------------------------------------------------------
// 10) Sequential greedy scan (matches the fori_loop semantics), then take the
//     first POST_NMS kept boxes (scores are sorted) and zero-fill the rest.
// ---------------------------------------------------------------------------
__global__ __launch_bounds__(256) void k_nms_scan(const unsigned* __restrict__ mask,
                                                  const float* __restrict__ sscore,
                                                  const float* __restrict__ sboxes,
                                                  float* __restrict__ rois) {
  int b = blockIdx.x;
  int tid = threadIdx.x;
  __shared__ unsigned remove[NWORDS];
  __shared__ int af;
  if (tid < NWORDS) remove[tid] = 0u;
  __syncthreads();
  for (int i = 0; i < PRE_NMS; ++i) {
    if (tid == 0) {
      bool fin = sscore[(size_t)b * PRE_NMS + i] > -1e37f;
      bool rem = (remove[i >> 5] >> (i & 31)) & 1u;
      af = (fin && !rem) ? 1 : 0;
    }
    __syncthreads();
    if (af && tid < NWORDS)
      remove[tid] |= mask[((size_t)b * PRE_NMS + i) * NWORDS + tid];
    __syncthreads();
  }
  if (tid == 0) {
    float* ro = rois + (size_t)b * POST_NMS * 4;
    int cnt = 0;
    for (int i = 0; i < PRE_NMS && cnt < POST_NMS; ++i) {
      bool fin = sscore[(size_t)b * PRE_NMS + i] > -1e37f;
      bool rem = (remove[i >> 5] >> (i & 31)) & 1u;
      if (fin && !rem) {
        const float* p = sboxes + ((size_t)b * PRE_NMS + i) * 4;
        ro[cnt * 4 + 0] = p[0]; ro[cnt * 4 + 1] = p[1];
        ro[cnt * 4 + 2] = p[2]; ro[cnt * 4 + 3] = p[3];
        ++cnt;
      }
    }
    for (; cnt < POST_NMS; ++cnt) {
      ro[cnt * 4 + 0] = 0.0f; ro[cnt * 4 + 1] = 0.0f;
      ro[cnt * 4 + 2] = 0.0f; ro[cnt * 4 + 3] = 0.0f;
    }
  }
}

// ---------------------------------------------------------------------------
extern "C" void kernel_launch(void* const* d_in, const int* in_sizes, int n_in,
                              void* d_out, int out_size, void* d_ws, size_t ws_size,
                              hipStream_t stream) {
  const float* x        = (const float*)d_in[0];
  const float* conv1_w  = (const float*)d_in[1];
  const float* conv1_b  = (const float*)d_in[2];
  const float* score_w  = (const float*)d_in[3];
  const float* score_b  = (const float*)d_in[4];
  const float* loc_w    = (const float*)d_in[5];
  const float* loc_b    = (const float*)d_in[6];
  const int*   img_h    = (const int*)d_in[7];
  const int*   img_w    = (const int*)d_in[8];

  float* out        = (float*)d_out;
  float* out_locs   = out;                                   // 8*34200*4
  float* out_scores = out + (size_t)BATCH * NPIX * 4;        // 8*34200*2
  float* out_rois   = out_scores + (size_t)BATCH * NPIX * 2; // 8*300*4
  float* out_anchor = out_rois + (size_t)BATCH * POST_NMS * 4; // 34200*4

  char* ws = (char*)d_ws;
  size_t off = 0;
  auto carve = [&](size_t bytes) -> char* {
    char* p = ws + off;
    off += bytes;
    off = (off + 255) & ~((size_t)255);
    return p;
  };
  _Float16* xb       = (_Float16*)carve((size_t)BATCH * PH * PW * CIN * 2);
  _Float16* feat     = (_Float16*)carve((size_t)MROWS * CIN * 2);
  _Float16* wpack    = (_Float16*)carve((size_t)9 * 16 * 32 * 32 * 16 * 2);
  _Float16* hpack    = (_Float16*)carve((size_t)16 * 4 * 32 * 16 * 2);
  float* boxes       = (float*)carve((size_t)BATCH * NPIX * 4 * 4);
  float* score_ws    = (float*)carve((size_t)BATCH * NPIX * 4);
  unsigned long long* keys = (unsigned long long*)carve((size_t)BATCH * SORTN * 8);
  float* sboxes      = (float*)carve((size_t)BATCH * PRE_NMS * 4 * 4);
  float* sarea       = (float*)carve((size_t)BATCH * PRE_NMS * 4);
  float* sscore      = (float*)carve((size_t)BATCH * PRE_NMS * 4);
  unsigned* mask     = (unsigned*)carve((size_t)BATCH * PRE_NMS * NWORDS * 4);

  // 1) layout transforms + weight packing
  {
    int n = BATCH * PH * PW * CIN;
    k_x_to_nhwc<<<(n + 255) / 256, 256, 0, stream>>>(x, xb);
  }
  k_pack_w1<<<(9 * 16 * 32 * 32 + 255) / 256, 256, 0, stream>>>(conv1_w, wpack);
  k_pack_whead<<<(16 * 4 * 32 + 255) / 256, 256, 0, stream>>>(loc_w, score_w, hpack);

  // 2) conv1 (WMMA implicit GEMM) + head GEMM
  k_conv1<<<dim3(MROWS / 64, CIN / 256, 1), 256, 0, stream>>>(xb, wpack, conv1_b, feat);
  k_head<<<MROWS / 64, 128, 0, stream>>>(feat, hpack, loc_b, score_b, out_locs, out_scores);

  // 3) proposals
  k_boxes<<<(BATCH * SORTN + 255) / 256, 256, 0, stream>>>(
      out_locs, out_scores, img_h, img_w, boxes, score_ws, keys, out_anchor);

  for (int k = 2; k <= SORTN; k <<= 1)
    for (int j = k >> 1; j > 0; j >>= 1)
      k_bitonic<<<(BATCH * SORTN + 255) / 256, 256, 0, stream>>>(keys, k, j);

  k_gather<<<(BATCH * PRE_NMS + 255) / 256, 256, 0, stream>>>(
      keys, boxes, score_ws, sboxes, sarea, sscore);

  k_nms_mask<<<dim3((PRE_NMS + 127) / 128, NWORDS, BATCH), 128, 0, stream>>>(
      sboxes, sarea, mask);

  k_nms_scan<<<BATCH, 256, 0, stream>>>(mask, sscore, sboxes, out_rois);
}
